// Decoder_13254269075775
// MI455X (gfx1250) — matrix operand order repaired and checked by reading, not compiled, not run on hardware
//
#include <hip/hip_runtime.h>
#include <math.h>

#define B_    64
#define T_ENC 1152
#define T_DEC 256
#define MEL   80
#define ENC_D 256
#define ATTN_D 128
#define PRE_D 256
#define RNN_D 1024
#define NGATE 4096          // 4*RNN_D
#define KX    1280          // ATTN_D + T_ENC
#define NT_N  (NGATE/16)    // 256 N tiles
#define KT_X  (KX/32)       // 40 K tiles (x path)
#define KT_H  (RNN_D/32)    // 32 K tiles (h path)
#define MT    (B_/16)       // 4 M tiles

typedef __attribute__((ext_vector_type(16))) __bf16 v16bf;
typedef __attribute__((ext_vector_type(8)))  float  v8f;

// ---------------- workspace layout (bytes) ----------------
#define WS_PM    0u                                   // f32 [B,T_ENC,ATTN_D]  37,748,736
#define WS_WIH   (WS_PM   + 37748736u)                // bf16 packed W_ih      10,485,760
#define WS_WHH   (WS_WIH  + 10485760u)                // bf16 packed W_hh       8,388,608
#define WS_XA    (WS_WHH  + 8388608u)                 // bf16 packed x A-tiles    163,840
#define WS_HA    (WS_XA   + 163840u)                  // bf16 packed h A-tiles    131,072
#define WS_XBUF  (WS_HA   + 131072u)                  // f32 [B,KX]               327,680
#define WS_HBUF  (WS_XBUF + 327680u)                  // f32 [B,RNN_D]            262,144
#define WS_CBUF  (WS_HBUF + 262144u)                  // f32 [B,RNN_D]            262,144
#define WS_GATES (WS_CBUF + 262144u)                  // f32 [B,NGATE]          1,048,576

// ---------------- output layout (floats) ----------------
#define OUT_MELS   0u
#define OUT_GATES  (OUT_MELS + (unsigned)(B_*T_DEC*MEL))     // 1,310,720
#define OUT_ALIGNS (OUT_GATES + (unsigned)(B_*T_DEC))        // 1,327,104

__device__ __forceinline__ float sigm(float x) { return 1.0f / (1.0f + expf(-x)); }

// -------- once per launch: projected memory pm[b,t,d] = enc[b,t,:] @ Wm[:,d] --------
__global__ void k_pm(const float* __restrict__ enc, const float* __restrict__ Wm,
                     float* __restrict__ pm) {
    int bt = blockIdx.x;           // B*T_ENC blocks, 128 threads (d)
    int d  = threadIdx.x;
    __shared__ float se[ENC_D];
    const float* row = enc + (size_t)bt * ENC_D;
    for (int e = d; e < ENC_D; e += 128) se[e] = row[e];
    __syncthreads();
    float acc = 0.0f;
    for (int e = 0; e < ENC_D; ++e) acc += se[e] * Wm[e * ATTN_D + d];
    pm[(size_t)bt * ATTN_D + d] = acc;
}

// -------- pack f32 weight [Kdim, NGATE] into bf16 WMMA B-operand lane layout --------
// tile index = kt*NT_N + nt ; lane<16: K = kt*32 + i ; lane>=16: K = kt*32 + 16 + i
__global__ void k_packW(const float* __restrict__ W, __bf16* __restrict__ out, int nKt) {
    int gid  = blockIdx.x * 256 + threadIdx.x;
    int lane = gid & 31;
    int tile = gid >> 5;
    if (tile >= nKt * NT_N) return;
    int kt = tile / NT_N, nt = tile % NT_N;
    int n = nt * 16 + (lane & 15);
    int kbase = kt * 32 + ((lane < 16) ? 0 : 16);
    __bf16* dst = out + ((size_t)tile * 32 + lane) * 16;
#pragma unroll
    for (int i = 0; i < 16; ++i)
        dst[i] = (__bf16)W[(size_t)(kbase + i) * NGATE + n];
}

__global__ void k_zero(float* __restrict__ p, int n) {
    int i = blockIdx.x * 256 + threadIdx.x;
    if (i < n) p[i] = 0.0f;
}

// -------- per step (fused): prenet -> query -> attention -> softmax -> context --------
// One block per batch row; emits aligns output and the LSTM input row [ctx || w].
__global__ void k_front(const float* __restrict__ mel_outputs, int t,
                        const float* __restrict__ W1, const float* __restrict__ b1,
                        const float* __restrict__ W2, const float* __restrict__ b2,
                        const float* __restrict__ Wq,
                        const float* __restrict__ pm, const float* __restrict__ attn_v,
                        float* __restrict__ xbuf, float* __restrict__ out) {
    int b = blockIdx.x, tid = threadIdx.x;                 // 256 threads
    __shared__ float sx[MEL];
    __shared__ float sp1[PRE_D];
    __shared__ float sp2[PRE_D];
    __shared__ float sq[ATTN_D], sv[ATTN_D];
    __shared__ float se[T_ENC];
    __shared__ float sred[256];

    // ---- prenet ----
    if (tid < MEL)
        sx[tid] = (t == 0) ? 0.0f : mel_outputs[((size_t)b * T_DEC + t) * MEL + tid];
    if (tid < ATTN_D) sv[tid] = attn_v[tid];
    __syncthreads();
    float a1 = b1[tid];
    for (int m = 0; m < MEL; ++m) a1 += sx[m] * W1[m * PRE_D + tid];
    sp1[tid] = fmaxf(a1, 0.0f);
    __syncthreads();
    float a2 = b2[tid];
    for (int m = 0; m < PRE_D; ++m) a2 += sp1[m] * W2[m * PRE_D + tid];
    sp2[tid] = fmaxf(a2, 0.0f);
    __syncthreads();
    // ---- query projection ----
    if (tid < ATTN_D) {
        float q = 0.0f;
        for (int k = 0; k < PRE_D; ++k) q += sp2[k] * Wq[k * ATTN_D + tid];
        sq[tid] = q;
    }
    __syncthreads();
    // ---- energies ----
    const float* pmb = pm + (size_t)b * T_ENC * ATTN_D;
    for (int tt = tid; tt < T_ENC; tt += 256) {
        const float* row = pmb + (size_t)tt * ATTN_D;
        float acc = 0.0f;
        for (int d = 0; d < ATTN_D; ++d) acc += tanhf(sq[d] + row[d]) * sv[d];
        se[tt] = acc;
    }
    __syncthreads();
    // ---- block max ----
    float mx = -1e30f;
    for (int tt = tid; tt < T_ENC; tt += 256) mx = fmaxf(mx, se[tt]);
    sred[tid] = mx; __syncthreads();
    for (int s = 128; s > 0; s >>= 1) { if (tid < s) sred[tid] = fmaxf(sred[tid], sred[tid + s]); __syncthreads(); }
    mx = sred[0]; __syncthreads();
    // ---- exp + block sum ----
    float sum = 0.0f;
    for (int tt = tid; tt < T_ENC; tt += 256) { float e = expf(se[tt] - mx); se[tt] = e; sum += e; }
    sred[tid] = sum; __syncthreads();
    for (int s = 128; s > 0; s >>= 1) { if (tid < s) sred[tid] += sred[tid + s]; __syncthreads(); }
    float inv = 1.0f / sred[0]; __syncthreads();
    // ---- normalize, emit alignment and LSTM-input tail ----
    float* aligns = out + OUT_ALIGNS + ((size_t)b * T_DEC + t) * T_ENC;
    for (int tt = tid; tt < T_ENC; tt += 256) {
        float w = se[tt] * inv;
        se[tt] = w;
        aligns[tt] = w;
        xbuf[(size_t)b * KX + ATTN_D + tt] = w;
    }
    __syncthreads();
    // ---- context over projected memory (coalesced across d) ----
    if (tid < ATTN_D) {
        float acc = 0.0f;
        for (int tt = 0; tt < T_ENC; ++tt) acc += se[tt] * pmb[(size_t)tt * ATTN_D + tid];
        xbuf[(size_t)b * KX + tid] = acc;
    }
}

// -------- per step: pack x=[ctx||w] and h into bf16 WMMA A-operand lane layout --------
// A tile index = mt*nKt + kt ; lane<16: m=mt*16+lane, K in {0..15}; lane>=16: K in {16..31}
__global__ void k_packA(const float* __restrict__ xbuf, const float* __restrict__ hbuf,
                        __bf16* __restrict__ xA, __bf16* __restrict__ hA) {
    int gid  = blockIdx.x * 256 + threadIdx.x;
    int lane = gid & 31;
    int tile = gid >> 5;
    const int XT = MT * KT_X, HT = MT * KT_H;
    if (tile >= XT + HT) return;
    const float* src; __bf16* dst; int nKt, Kdim, lt;
    if (tile < XT) { nKt = KT_X; Kdim = KX;    lt = tile;      src = xbuf; dst = xA; }
    else           { nKt = KT_H; Kdim = RNN_D; lt = tile - XT; src = hbuf; dst = hA; }
    int mt = lt / nKt, kt = lt % nKt;
    int m = mt * 16 + (lane & 15);
    int khalf = (lane < 16) ? 0 : 16;
    const float* row = src + (size_t)m * Kdim + kt * 32;
    __bf16* o = dst + ((size_t)lt * 32 + lane) * 16;
#pragma unroll
    for (int i = 0; i < 16; ++i) {
        int k = khalf + i;
        o[i] = (__bf16)row[k];
    }
}

// -------- per step: gates = x@W_ih + h@W_hh via v_wmma_f32_16x16x32_bf16 --------
// Each wave computes two adjacent N-tiles so every A fetch feeds two WMMAs.
__global__ void k_gemm(const __bf16* __restrict__ xA, const __bf16* __restrict__ Wih,
                       const __bf16* __restrict__ hA, const __bf16* __restrict__ Whh,
                       float* __restrict__ gates) {
    int wave = threadIdx.x >> 5;                // 128 threads = 4 waves / block
    int lane = threadIdx.x & 31;
    int tileId = blockIdx.x * 4 + wave;         // 128 blocks -> 512 wave-tiles
    int mt  = tileId >> 7;                      // / 128 pairs
    int nt0 = (tileId & 127) * 2;
    v8f acc0 = {}, acc1 = {};
#pragma unroll 2
    for (int kt = 0; kt < KT_X; ++kt) {
        v16bf a  = *(const v16bf*)(xA  + ((size_t)(mt * KT_X + kt) * 32 + lane) * 16);
        v16bf b0 = *(const v16bf*)(Wih + ((size_t)(kt * NT_N + nt0    ) * 32 + lane) * 16);
        v16bf b1 = *(const v16bf*)(Wih + ((size_t)(kt * NT_N + nt0 + 1) * 32 + lane) * 16);
        acc0 = __builtin_amdgcn_wmma_f32_16x16x32_bf16(false, a, false, b0, (short)0, acc0, false, false);
        acc1 = __builtin_amdgcn_wmma_f32_16x16x32_bf16(false, a, false, b1, (short)0, acc1, false, false);
    }
#pragma unroll 2
    for (int kt = 0; kt < KT_H; ++kt) {
        v16bf a  = *(const v16bf*)(hA  + ((size_t)(mt * KT_H + kt) * 32 + lane) * 16);
        v16bf b0 = *(const v16bf*)(Whh + ((size_t)(kt * NT_N + nt0    ) * 32 + lane) * 16);
        v16bf b1 = *(const v16bf*)(Whh + ((size_t)(kt * NT_N + nt0 + 1) * 32 + lane) * 16);
        acc0 = __builtin_amdgcn_wmma_f32_16x16x32_bf16(false, a, false, b0, (short)0, acc0, false, false);
        acc1 = __builtin_amdgcn_wmma_f32_16x16x32_bf16(false, a, false, b1, (short)0, acc1, false, false);
    }
    // f32 16x16 C/D layout: VGPR r, lanes 0-15 -> M=r, lanes 16-31 -> M=8+r
    int n0 = nt0 * 16 + (lane & 15);
    int mbase = mt * 16 + ((lane < 16) ? 0 : 8);
#pragma unroll
    for (int r = 0; r < 8; ++r) {
        gates[(size_t)(mbase + r) * NGATE + n0]      = acc0[r];
        gates[(size_t)(mbase + r) * NGATE + n0 + 16] = acc1[r];
    }
}

// -------- per step (fused): LSTM pointwise (i,f,g,o) + mel / stop projections --------
// One 1024-thread block per batch row; h lands in LDS for the output dots.
__global__ void k_back(const float* __restrict__ gates,
                       const float* __restrict__ b_ih, const float* __restrict__ b_hh,
                       float* __restrict__ h, float* __restrict__ c,
                       const float* __restrict__ melW, const float* __restrict__ melB,
                       const float* __restrict__ stopW, const float* __restrict__ stopB,
                       float* __restrict__ out, int t) {
    int b = blockIdx.x, n = threadIdx.x;        // 1024 threads
    __shared__ float sh[RNN_D];
    const float* g = gates + (size_t)b * NGATE;
    float iv = g[n]            + b_ih[n]            + b_hh[n];
    float fv = g[RNN_D + n]    + b_ih[RNN_D + n]    + b_hh[RNN_D + n];
    float gv = g[2*RNN_D + n]  + b_ih[2*RNN_D + n]  + b_hh[2*RNN_D + n];
    float ov = g[3*RNN_D + n]  + b_ih[3*RNN_D + n]  + b_hh[3*RNN_D + n];
    int idx = b * RNN_D + n;
    float cn = sigm(fv) * c[idx] + sigm(iv) * tanhf(gv);
    float hn = sigm(ov) * tanhf(cn);
    c[idx] = cn;
    h[idx] = hn;
    sh[n] = hn;
    __syncthreads();
    if (n < MEL) {
        float acc = melB[n];
        for (int k = 0; k < RNN_D; ++k) acc += sh[k] * melW[k * MEL + n];
        out[OUT_MELS + ((size_t)b * T_DEC + t) * MEL + n] = acc;
    } else if (n == MEL) {
        float acc = stopB[0];
        for (int k = 0; k < RNN_D; ++k) acc += sh[k] * stopW[k];
        out[OUT_GATES + (size_t)b * T_DEC + t] = sigm(acc);
    }
}

extern "C" void kernel_launch(void* const* d_in, const int* in_sizes, int n_in,
                              void* d_out, int out_size, void* d_ws, size_t ws_size,
                              hipStream_t stream) {
    (void)in_sizes; (void)n_in; (void)out_size; (void)ws_size;
    const float* enc     = (const float*)d_in[0];
    const float* mel_out = (const float*)d_in[1];
    // d_in[2], d_in[3]: text/mel lengths (unused by reference math)
    const float* pW1 = (const float*)d_in[4];
    const float* pb1 = (const float*)d_in[5];
    const float* pW2 = (const float*)d_in[6];
    const float* pb2 = (const float*)d_in[7];
    const float* Wq  = (const float*)d_in[8];
    const float* Wm  = (const float*)d_in[9];
    const float* av  = (const float*)d_in[10];
    const float* Wih = (const float*)d_in[11];
    const float* Whh = (const float*)d_in[12];
    const float* bih = (const float*)d_in[13];
    const float* bhh = (const float*)d_in[14];
    const float* mW  = (const float*)d_in[15];
    const float* mB  = (const float*)d_in[16];
    const float* sW  = (const float*)d_in[17];
    const float* sB  = (const float*)d_in[18];

    char* ws = (char*)d_ws;
    float*  pm    = (float*)(ws + WS_PM);
    __bf16* WihP  = (__bf16*)(ws + WS_WIH);
    __bf16* WhhP  = (__bf16*)(ws + WS_WHH);
    __bf16* xA    = (__bf16*)(ws + WS_XA);
    __bf16* hA    = (__bf16*)(ws + WS_HA);
    float*  xbuf  = (float*)(ws + WS_XBUF);
    float*  hbuf  = (float*)(ws + WS_HBUF);
    float*  cbuf  = (float*)(ws + WS_CBUF);
    float*  gates = (float*)(ws + WS_GATES);
    float*  out   = (float*)d_out;

    // once per launch: projected memory, bf16 weight packing, state init
    k_pm<<<B_ * T_ENC, 128, 0, stream>>>(enc, Wm, pm);
    k_packW<<<(KT_X * NT_N * 32 + 255) / 256, 256, 0, stream>>>(Wih, WihP, KT_X);
    k_packW<<<(KT_H * NT_N * 32 + 255) / 256, 256, 0, stream>>>(Whh, WhhP, KT_H);
    k_zero<<<(B_ * RNN_D + 255) / 256, 256, 0, stream>>>(hbuf, B_ * RNN_D);
    k_zero<<<(B_ * RNN_D + 255) / 256, 256, 0, stream>>>(cbuf, B_ * RNN_D);

    for (int t = 0; t < T_DEC; ++t) {
        k_front<<<B_, 256, 0, stream>>>(mel_out, t, pW1, pb1, pW2, pb2, Wq, pm, av, xbuf, out);
        k_packA<<<(MT * (KT_X + KT_H) * 32 + 255) / 256, 256, 0, stream>>>(xbuf, hbuf, xA, hA);
        k_gemm<<<(MT * NT_N / 2) / 4, 128, 0, stream>>>(xA, WihP, hA, WhhP, gates);
        k_back<<<B_, 1024, 0, stream>>>(gates, bih, bhh, hbuf, cbuf, mW, mB, sW, sB, out, t);
    }
}
